// MXLlamaDecoderLayer_85770496901392
// MI455X (gfx1250) — 8-bit matrix operand order repaired and checked by reading, compiled, not run on hardware
//
#include <hip/hip_runtime.h>
#include <hip/hip_bf16.h>

#define SEQ   1024
#define HDIM  4096
#define NHEAD 32
#define DHEAD 128
#define FFDIM 11008

typedef __attribute__((ext_vector_type(16))) int      v16i;
typedef __attribute__((ext_vector_type(8)))  float    v8f;
typedef __attribute__((ext_vector_type(4)))  unsigned v4u;
typedef __attribute__((ext_vector_type(8)))  int      v8i_t;
typedef __attribute__((ext_vector_type(4)))  int      v4i_t;
typedef int b128_t __attribute__((vector_size(16)));

#if __has_builtin(__builtin_amdgcn_tensor_load_to_lds) && \
    __has_builtin(__builtin_amdgcn_s_wait_tensorcnt)
#define USE_TDM 1
#else
#define USE_TDM 0
#endif

#if __has_builtin(__builtin_amdgcn_global_load_async_to_lds_b128) && \
    __has_builtin(__builtin_amdgcn_s_wait_asynccnt)
#define USE_ASYNC_LDS 1
#define ASYNC_CP16(ldsp, gp, off)                                          \
  __builtin_amdgcn_global_load_async_to_lds_b128(                          \
      (__attribute__((address_space(1))) b128_t*)(gp),                     \
      (__attribute__((address_space(3))) b128_t*)(ldsp), (off), 0)
#define WAIT_ASYNC(n) __builtin_amdgcn_s_wait_asynccnt(n)
#else
#define USE_ASYNC_LDS 0
#endif

#if USE_TDM
__device__ __forceinline__ void tdm_load_tile(const unsigned char* gbase,
                                              unsigned lds_off, unsigned kBytes)
{
  unsigned long long ga = (unsigned long long)(uintptr_t)gbase;
  v4u g0;
  g0[0] = 1u;
  g0[1] = lds_off;
  g0[2] = (unsigned)(ga & 0xFFFFFFFFu);
  g0[3] = (unsigned)((ga >> 32) & 0x1FFFFFFu)
        | (2u << 30);
  v8i_t g1;
  g1[0] = (int)((1u << 20) | (4u << 22) | (3u << 25));
  g1[1] = (int)((kBytes & 0xFFFFu) << 16);
  g1[2] = (int)(((kBytes >> 16) & 0xFFFFu) | (128u << 16));
  g1[3] = (int)(128u << 16);
  g1[4] = (int)128u;
  g1[5] = (int)kBytes;
  g1[6] = 0;
  g1[7] = 0;
  v4i_t g2; g2[0] = 1; g2[1] = 0; g2[2] = 0; g2[3] = 0;
  v4i_t g3; g3[0] = 0; g3[1] = 0; g3[2] = 0; g3[3] = 0;
  v8i_t g4; g4[0] = 0; g4[1] = 0; g4[2] = 0; g4[3] = 0;
            g4[4] = 0; g4[5] = 0; g4[6] = 0; g4[7] = 0;
  __builtin_amdgcn_tensor_load_to_lds(g0, g1, g2, g3, g4, 0);
}
#define LDS_OFFSET(p) ((unsigned)(uintptr_t)(__attribute__((address_space(3))) unsigned char*)(p))
#endif

__device__ __forceinline__ float wredmax(float v) {
#pragma unroll
  for (int o = 16; o > 0; o >>= 1) v = fmaxf(v, __shfl_xor(v, o, 32));
  return v;
}
__device__ __forceinline__ float wredsum(float v) {
#pragma unroll
  for (int o = 16; o > 0; o >>= 1) v += __shfl_xor(v, o, 32);
  return v;
}

__device__ __forceinline__ float mx_scale(float am) {
  if (!(am > 1e-30f)) return 1.0f;
  int e = (int)(__float_as_uint(am) >> 23) - 127;
  return exp2f((float)(e - 8));
}

__device__ __forceinline__ unsigned q_e4m3(float v, float inv) {
  float x = v * inv;
  unsigned sgn = (__float_as_uint(x) >> 24) & 0x80u;
  float ax = fabsf(x);
  if (!(ax > 0.f)) return sgn;
  ax = fminf(ax, 448.0f);
  if (ax < 0.015625f) {
    int m = (int)rintf(ax * 512.0f);
    if (m > 7) return sgn | 0x08u;
    return sgn | (unsigned)m;
  }
  int e = (int)(__float_as_uint(ax) >> 23) - 127;
  float sc = exp2f((float)(e - 3));
  int qm = (int)rintf(ax / sc);
  if (qm == 16) { e += 1; qm = 8; }
  if (e > 8) { e = 8; qm = 14; }
  return sgn | (unsigned)((e + 7) << 3) | (unsigned)(qm - 8);
}

__device__ __forceinline__ unsigned pack4_e4m3(float a, float b, float c, float d, float inv) {
  return q_e4m3(a, inv) | (q_e4m3(b, inv) << 8) | (q_e4m3(c, inv) << 16) | (q_e4m3(d, inv) << 24);
}

__global__ __launch_bounds__(256)
void k_gemm_fp8(const unsigned char* __restrict__ A, const float* __restrict__ sA,
                long long aBatch, long long saBatch,
                const unsigned char* __restrict__ B, const float* __restrict__ sB,
                long long bBatch, long long sbBatch,
                float* __restrict__ C, long long cBatch,
                int M, int N, int K)
{
#if USE_TDM || USE_ASYNC_LDS
  __shared__ unsigned char lA[2][128 * 144];
  __shared__ unsigned char lB[2][128 * 144];
  __shared__ float lsA[2][128];
  __shared__ float lsB[2][128];
#else
  __shared__ unsigned char lA[1][128 * 144];
  __shared__ unsigned char lB[1][128 * 144];
  __shared__ float lsA[1][128];
  __shared__ float lsB[1][128];
#endif

  const int z = blockIdx.z;
  A  += (size_t)z * aBatch;   B  += (size_t)z * bBatch;   C += (size_t)z * cBatch;
  sA += (size_t)z * saBatch;  sB += (size_t)z * sbBatch;

  const int KC   = K >> 7;
  const int bm   = blockIdx.x * 128, bn = blockIdx.y * 128;
  const int tid  = threadIdx.x;
  const int wave = tid >> 5, lane = tid & 31;
  const int wm   = (wave & 1) * 64;
  const int wn   = (wave >> 1) * 32;
  const int lrow  = tid >> 1;
  const int lhalf = (tid & 1) * 64;
  (void)lrow; (void)lhalf;

  const v8f zero8 = {0.f, 0.f, 0.f, 0.f, 0.f, 0.f, 0.f, 0.f};
  v8f acc[4][2];
#pragma unroll
  for (int mt = 0; mt < 4; ++mt)
#pragma unroll
    for (int nt = 0; nt < 2; ++nt) acc[mt][nt] = zero8;

  auto stage_scales = [&](int kc, int buf) {
    if (tid < 128) lsA[buf][tid]       = sA[(size_t)(bm + tid) * KC + kc];
    else           lsB[buf][tid - 128] = sB[(size_t)(bn + tid - 128) * KC + kc];
  };

#if USE_TDM
  auto tdm_issue = [&](int kc, int buf) {
    tdm_load_tile(A + (size_t)bm * K + (size_t)kc * 128, LDS_OFFSET(&lA[buf][0]), (unsigned)K);
    tdm_load_tile(B + (size_t)bn * K + (size_t)kc * 128, LDS_OFFSET(&lB[buf][0]), (unsigned)K);
  };
  if (wave == 0) tdm_issue(0, 0);
#elif USE_ASYNC_LDS
  auto stage_async = [&](int kc, int buf) {
    const unsigned char* ga = A + (size_t)(bm + lrow) * K + (size_t)kc * 128 + lhalf;
    unsigned char* la = &lA[buf][lrow * 144 + lhalf];
    ASYNC_CP16(la, ga, 0);  ASYNC_CP16(la, ga, 16);
    ASYNC_CP16(la, ga, 32); ASYNC_CP16(la, ga, 48);
    const unsigned char* gb = B + (size_t)(bn + lrow) * K + (size_t)kc * 128 + lhalf;
    unsigned char* lb = &lB[buf][lrow * 144 + lhalf];
    ASYNC_CP16(lb, gb, 0);  ASYNC_CP16(lb, gb, 16);
    ASYNC_CP16(lb, gb, 32); ASYNC_CP16(lb, gb, 48);
    stage_scales(kc, buf);
  };
  stage_async(0, 0);
#endif

  for (int kc = 0; kc < KC; ++kc) {
#if USE_TDM
    const int cur = kc & 1;
    if (wave == 0) {
      if (kc + 1 < KC) {
        tdm_issue(kc + 1, cur ^ 1);
        __builtin_amdgcn_s_wait_tensorcnt(2);
      } else {
        __builtin_amdgcn_s_wait_tensorcnt(0);
      }
    }
    stage_scales(kc, cur);
    __syncthreads();
#elif USE_ASYNC_LDS
    const int cur = kc & 1;
    if (kc + 1 < KC) {
      stage_async(kc + 1, cur ^ 1);
      WAIT_ASYNC(8);
    } else {
      WAIT_ASYNC(0);
    }
    __syncthreads();
#else
    const int cur = 0;
    {
      const uint4* ga = (const uint4*)(A + (size_t)(bm + lrow) * K + (size_t)kc * 128 + lhalf);
      uint4* la = (uint4*)(&lA[0][lrow * 144 + lhalf]);
      la[0] = ga[0]; la[1] = ga[1]; la[2] = ga[2]; la[3] = ga[3];
      const uint4* gb = (const uint4*)(B + (size_t)(bn + lrow) * K + (size_t)kc * 128 + lhalf);
      uint4* lb = (uint4*)(&lB[0][lrow * 144 + lhalf]);
      lb[0] = gb[0]; lb[1] = gb[1]; lb[2] = gb[2]; lb[3] = gb[3];
      stage_scales(kc, 0);
      if (kc + 1 < KC) {
        __builtin_prefetch(A + (size_t)(bm + lrow) * K + (size_t)(kc + 1) * 128 + lhalf, 0, 0);
        __builtin_prefetch(B + (size_t)(bn + lrow) * K + (size_t)(kc + 1) * 128 + lhalf, 0, 0);
      }
    }
    __syncthreads();
#endif

    v16i af[4];
#pragma unroll
    for (int mt = 0; mt < 4; ++mt) {
      const int arow = wm + mt * 16 + (lane & 15);
      const unsigned long long* ap =
          (const unsigned long long*)(&lA[cur][arow * 144 + ((lane >> 4) << 4)]);
#pragma unroll
      for (int j = 0; j < 8; ++j) {
        unsigned long long q = ap[4 * (j >> 1) + (j & 1)];
        af[mt][2 * j]     = (int)(unsigned)q;
        af[mt][2 * j + 1] = (int)(unsigned)(q >> 32);
      }
    }

#pragma unroll
    for (int nt = 0; nt < 2; ++nt) {
      const int bcol = wn + nt * 16 + (lane & 15);
      const uint4* bp = (const uint4*)(&lB[cur][bcol * 144 + ((lane >> 4) << 4)]);
      v16i bf;
#pragma unroll
      for (int j = 0; j < 4; ++j) {
        uint4 d = bp[2 * j];
        bf[4 * j + 0] = (int)d.x; bf[4 * j + 1] = (int)d.y;
        bf[4 * j + 2] = (int)d.z; bf[4 * j + 3] = (int)d.w;
      }
      const float sb = lsB[cur][bcol];

#pragma unroll
      for (int mt = 0; mt < 4; ++mt) {
        v8f cz = zero8;
        v8f dq = __builtin_amdgcn_wmma_f32_16x16x128_fp8_fp8(af[mt], bf, (short)0, cz, false, false);
        const int mb = wm + mt * 16 + ((lane >> 4) << 3);
#pragma unroll
        for (int i = 0; i < 8; ++i) acc[mt][nt][i] += dq[i] * (lsA[cur][mb + i] * sb);
      }
    }
    __syncthreads();
  }

#pragma unroll
  for (int mt = 0; mt < 4; ++mt) {
    const int r0 = bm + wm + mt * 16 + ((lane >> 4) << 3);
#pragma unroll
    for (int nt = 0; nt < 2; ++nt) {
      const int cb = bn + wn + nt * 16 + (lane & 15);
#pragma unroll
      for (int i = 0; i < 8; ++i) C[(size_t)(r0 + i) * N + cb] = acc[mt][nt][i];
    }
  }
}

__global__ __launch_bounds__(256)
void k_rmsnorm(const float* __restrict__ x, const float* __restrict__ w,
               float* __restrict__ y, int C)
{
  const int row = blockIdx.x, tid = threadIdx.x;
  __shared__ float red[8];
  __shared__ float stot;
  const float* xr = x + (size_t)row * C;
  float ss = 0.f;
  for (int c = tid; c < C; c += 256) { float v = xr[c]; ss += v * v; }
  ss = wredsum(ss);
  if ((tid & 31) == 0) red[tid >> 5] = ss;
  __syncthreads();
  if (tid == 0) {
    float t = 0.f;
    for (int i = 0; i < 8; ++i) t += red[i];
    stot = rsqrtf(t / (float)C + 1e-5f);
  }
  __syncthreads();
  const float inv = stot;
  float* yr = y + (size_t)row * C;
  for (int c = tid; c < C; c += 256) yr[c] = xr[c] * inv * w[c];
}

__global__ __launch_bounds__(256)
void k_quant_rows(const float* __restrict__ x, unsigned char* __restrict__ q,
                  float* __restrict__ s, int C)
{
  const int row = blockIdx.x, wave = threadIdx.x >> 5, lane = threadIdx.x & 31;
  const int G = C >> 7;
  const float* xr = x + (size_t)row * C;
  unsigned char* qr = q + (size_t)row * C;
  for (int g = wave; g < G; g += 8) {
    float4 f = ((const float4*)(xr + (size_t)g * 128))[lane];
    float am = fmaxf(fmaxf(fabsf(f.x), fabsf(f.y)), fmaxf(fabsf(f.z), fabsf(f.w)));
    am = wredmax(am);
    float sc = mx_scale(am), inv = 1.0f / sc;
    ((unsigned*)(qr + (size_t)g * 128))[lane] = pack4_e4m3(f.x, f.y, f.z, f.w, inv);
    if (lane == 0) s[(size_t)row * G + g] = sc;
  }
}

__global__ __launch_bounds__(256)
void k_rope(float* __restrict__ x, const float* __restrict__ cosT, const float* __restrict__ sinT)
{
  const int s = blockIdx.x;
  for (int i = threadIdx.x; i < NHEAD * 64; i += 256) {
    const int h = i >> 6, d = i & 63;
    const size_t base = (size_t)s * HDIM + (size_t)h * DHEAD;
    float a = x[base + d], b = x[base + d + 64];
    float c = cosT[(size_t)s * DHEAD + d], sn = sinT[(size_t)s * DHEAD + d];
    x[base + d]      = a * c - b * sn;
    x[base + d + 64] = b * c + a * sn;
  }
}

__global__ __launch_bounds__(256)
void k_quant_heads(const float* __restrict__ x, unsigned char* __restrict__ q,
                   float* __restrict__ s)
{
  const int sI = blockIdx.x, wave = threadIdx.x >> 5, lane = threadIdx.x & 31;
  for (int h = wave; h < NHEAD; h += 8) {
    float4 f = ((const float4*)(x + (size_t)sI * HDIM + (size_t)h * DHEAD))[lane];
    float am = fmaxf(fmaxf(fabsf(f.x), fabsf(f.y)), fmaxf(fabsf(f.z), fabsf(f.w)));
    am = wredmax(am);
    float sc = mx_scale(am), inv = 1.0f / sc;
    ((unsigned*)(q + ((size_t)h * SEQ + sI) * DHEAD))[lane] = pack4_e4m3(f.x, f.y, f.z, f.w, inv);
    if (lane == 0) s[(size_t)h * SEQ + sI] = sc;
  }
}

__global__ __launch_bounds__(256)
void k_quant_from_heads(const float* __restrict__ x, unsigned char* __restrict__ q,
                        float* __restrict__ s)
{
  const int sI = blockIdx.x, wave = threadIdx.x >> 5, lane = threadIdx.x & 31;
  for (int h = wave; h < NHEAD; h += 8) {
    float4 f = ((const float4*)(x + ((size_t)h * SEQ + sI) * DHEAD))[lane];
    float am = fmaxf(fmaxf(fabsf(f.x), fabsf(f.y)), fmaxf(fabsf(f.z), fabsf(f.w)));
    am = wredmax(am);
    float sc = mx_scale(am), inv = 1.0f / sc;
    ((unsigned*)(q + (size_t)sI * HDIM + (size_t)h * DHEAD))[lane] = pack4_e4m3(f.x, f.y, f.z, f.w, inv);
    if (lane == 0) s[(size_t)sI * NHEAD + h] = sc;
  }
}

__global__ __launch_bounds__(256)
void k_quant_vT(const float* __restrict__ v, unsigned char* __restrict__ q,
                float* __restrict__ s)
{
  const int r = blockIdx.x;
  const int h = r >> 7, d = r & 127;
  const int wave = threadIdx.x >> 5, lane = threadIdx.x & 31;
  for (int g = wave; g < SEQ / 128; g += 8) {
    float f0 = v[(size_t)(g * 128 + lane * 4 + 0) * HDIM + (size_t)h * DHEAD + d];
    float f1 = v[(size_t)(g * 128 + lane * 4 + 1) * HDIM + (size_t)h * DHEAD + d];
    float f2 = v[(size_t)(g * 128 + lane * 4 + 2) * HDIM + (size_t)h * DHEAD + d];
    float f3 = v[(size_t)(g * 128 + lane * 4 + 3) * HDIM + (size_t)h * DHEAD + d];
    float am = fmaxf(fmaxf(fabsf(f0), fabsf(f1)), fmaxf(fabsf(f2), fabsf(f3)));
    am = wredmax(am);
    float sc = mx_scale(am), inv = 1.0f / sc;
    ((unsigned*)(q + (size_t)r * SEQ + (size_t)g * 128))[lane] = pack4_e4m3(f0, f1, f2, f3, inv);
    if (lane == 0) s[(size_t)r * (SEQ / 128) + g] = sc;
  }
}

__global__ __launch_bounds__(256)
void k_softmax_quant(const float* __restrict__ sc, unsigned char* __restrict__ p8,
                     float* __restrict__ ps)
{
  const int row = blockIdx.x, h = blockIdx.y;
  const int tid = threadIdx.x, lane = tid & 31, wave = tid >> 5;
  __shared__ float red[8];
  __shared__ float bval;
  const float* rp = sc + ((size_t)h * SEQ + row) * SEQ;
  float4 f = ((const float4*)rp)[tid];
  const float scl = 0.08838834764831845f;
  float v0 = f.x * scl, v1 = f.y * scl, v2 = f.z * scl, v3 = f.w * scl;
  const int c0 = tid * 4;
  if (c0 + 0 > row) v0 = -1e30f;
  if (c0 + 1 > row) v1 = -1e30f;
  if (c0 + 2 > row) v2 = -1e30f;
  if (c0 + 3 > row) v3 = -1e30f;
  float mx = fmaxf(fmaxf(v0, v1), fmaxf(v2, v3));
  mx = wredmax(mx);
  if (lane == 0) red[wave] = mx;
  __syncthreads();
  if (tid == 0) {
    float t = red[0];
    for (int i = 1; i < 8; ++i) t = fmaxf(t, red[i]);
    bval = t;
  }
  __syncthreads();
  const float bm = bval;
  v0 = __expf(v0 - bm); v1 = __expf(v1 - bm); v2 = __expf(v2 - bm); v3 = __expf(v3 - bm);
  float sum = wredsum(v0 + v1 + v2 + v3);
  if (lane == 0) red[wave] = sum;
  __syncthreads();
  if (tid == 0) {
    float t = 0.f;
    for (int i = 0; i < 8; ++i) t += red[i];
    bval = t;
  }
  __syncthreads();
  const float inv = 1.0f / bval;
  v0 *= inv; v1 *= inv; v2 *= inv; v3 *= inv;
  float am = wredmax(fmaxf(fmaxf(v0, v1), fmaxf(v2, v3)));
  float qs = mx_scale(am), qi = 1.0f / qs;
  ((unsigned*)(p8 + ((size_t)h * SEQ + row) * SEQ + (size_t)wave * 128))[lane] =
      pack4_e4m3(v0, v1, v2, v3, qi);
  if (lane == 0) ps[((size_t)h * SEQ + row) * (SEQ / 128) + wave] = qs;
}

__global__ __launch_bounds__(256)
void k_add(const float* __restrict__ a, const float* __restrict__ b,
           float* __restrict__ o, size_t n)
{
  for (size_t i = (size_t)blockIdx.x * 256 + threadIdx.x; i < n; i += (size_t)gridDim.x * 256)
    o[i] = a[i] + b[i];
}

__global__ __launch_bounds__(256)
void k_silu_mul(const float* __restrict__ g, const float* __restrict__ u,
                float* __restrict__ o, size_t n)
{
  for (size_t i = (size_t)blockIdx.x * 256 + threadIdx.x; i < n; i += (size_t)gridDim.x * 256) {
    float gv = g[i];
    o[i] = (gv / (1.0f + __expf(-gv))) * u[i];
  }
}

extern "C" void kernel_launch(void* const* d_in, const int* in_sizes, int n_in,
                              void* d_out, int out_size, void* d_ws, size_t ws_size,
                              hipStream_t stream)
{
  (void)in_sizes; (void)n_in; (void)out_size; (void)ws_size;
  const float* hidden = (const float*)d_in[0];
  const float* cosT   = (const float*)d_in[1];
  const float* sinT   = (const float*)d_in[2];
  const float* ln1w = (const float*)d_in[4];
  const float* ln2w = (const float*)d_in[5];
  const float* Wq = (const float*)d_in[6];
  const float* Wk = (const float*)d_in[7];
  const float* Wv = (const float*)d_in[8];
  const float* Wo = (const float*)d_in[9];
  const float* Wg = (const float*)d_in[10];
  const float* Wu = (const float*)d_in[11];
  const float* Wd = (const float*)d_in[12];
  float* outp = (float*)d_out;

  char* ws = (char*)d_ws;
  size_t off = 0;
  auto alloc = [&](size_t b) -> char* {
    char* p = ws + off;
    off += (b + 255) & ~(size_t)255;
    return p;
  };

  unsigned char* wq8 = (unsigned char*)alloc((size_t)HDIM * HDIM);
  unsigned char* wk8 = (unsigned char*)alloc((size_t)HDIM * HDIM);
  unsigned char* wv8 = (unsigned char*)alloc((size_t)HDIM * HDIM);
  unsigned char* wo8 = (unsigned char*)alloc((size_t)HDIM * HDIM);
  unsigned char* wg8 = (unsigned char*)alloc((size_t)FFDIM * HDIM);
  unsigned char* wu8 = (unsigned char*)alloc((size_t)FFDIM * HDIM);
  unsigned char* wd8 = (unsigned char*)alloc((size_t)HDIM * FFDIM);
  float* wqs = (float*)alloc((size_t)HDIM * (HDIM / 128) * 4);
  float* wks = (float*)alloc((size_t)HDIM * (HDIM / 128) * 4);
  float* wvs = (float*)alloc((size_t)HDIM * (HDIM / 128) * 4);
  float* wos = (float*)alloc((size_t)HDIM * (HDIM / 128) * 4);
  float* wgs = (float*)alloc((size_t)FFDIM * (HDIM / 128) * 4);
  float* wus = (float*)alloc((size_t)FFDIM * (HDIM / 128) * 4);
  float* wds = (float*)alloc((size_t)HDIM * (FFDIM / 128) * 4);
  float* hn = (float*)alloc((size_t)SEQ * HDIM * 4);
  unsigned char* act8 = (unsigned char*)alloc((size_t)SEQ * HDIM);
  float* acts = (float*)alloc((size_t)SEQ * (HDIM / 128) * 4);
  unsigned char* aff8 = (unsigned char*)alloc((size_t)SEQ * FFDIM);
  float* affs = (float*)alloc((size_t)SEQ * (FFDIM / 128) * 4);
  float* qb = (float*)alloc((size_t)SEQ * HDIM * 4);
  float* kb = (float*)alloc((size_t)SEQ * HDIM * 4);
  float* vb = (float*)alloc((size_t)SEQ * HDIM * 4);
  unsigned char* qq8 = (unsigned char*)alloc((size_t)NHEAD * SEQ * DHEAD);
  float* qqs = (float*)alloc((size_t)NHEAD * SEQ * 4);
  unsigned char* kk8 = (unsigned char*)alloc((size_t)NHEAD * SEQ * DHEAD);
  float* kks = (float*)alloc((size_t)NHEAD * SEQ * 4);
  unsigned char* vt8 = (unsigned char*)alloc((size_t)NHEAD * DHEAD * SEQ);
  float* vts = (float*)alloc((size_t)NHEAD * DHEAD * (SEQ / 128) * 4);
  float* scores = (float*)alloc((size_t)NHEAD * SEQ * SEQ * 4);
  unsigned char* p8 = (unsigned char*)alloc((size_t)NHEAD * SEQ * SEQ);
  float* ps = (float*)alloc((size_t)NHEAD * SEQ * (SEQ / 128) * 4);
  float* ao = (float*)alloc((size_t)NHEAD * SEQ * DHEAD * 4);
  float* aout = (float*)alloc((size_t)SEQ * HDIM * 4);
  float* h1 = (float*)alloc((size_t)SEQ * HDIM * 4);
  float* gb_ = (float*)alloc((size_t)SEQ * FFDIM * 4);
  float* ub_ = (float*)alloc((size_t)SEQ * FFDIM * 4);
  float* mb_ = (float*)alloc((size_t)SEQ * FFDIM * 4);
  float* db_ = (float*)alloc((size_t)SEQ * HDIM * 4);

  k_quant_rows<<<HDIM, 256, 0, stream>>>(Wq, wq8, wqs, HDIM);
  k_quant_rows<<<HDIM, 256, 0, stream>>>(Wk, wk8, wks, HDIM);
  k_quant_rows<<<HDIM, 256, 0, stream>>>(Wv, wv8, wvs, HDIM);
  k_quant_rows<<<HDIM, 256, 0, stream>>>(Wo, wo8, wos, HDIM);
  k_quant_rows<<<FFDIM, 256, 0, stream>>>(Wg, wg8, wgs, HDIM);
  k_quant_rows<<<FFDIM, 256, 0, stream>>>(Wu, wu8, wus, HDIM);
  k_quant_rows<<<HDIM, 256, 0, stream>>>(Wd, wd8, wds, FFDIM);

  k_rmsnorm<<<SEQ, 256, 0, stream>>>(hidden, ln1w, hn, HDIM);
  k_quant_rows<<<SEQ, 256, 0, stream>>>(hn, act8, acts, HDIM);

  dim3 gqkv(SEQ / 128, HDIM / 128, 1);
  k_gemm_fp8<<<gqkv, 256, 0, stream>>>(act8, acts, 0, 0, wq8, wqs, 0, 0, qb, 0, SEQ, HDIM, HDIM);
  k_gemm_fp8<<<gqkv, 256, 0, stream>>>(act8, acts, 0, 0, wk8, wks, 0, 0, kb, 0, SEQ, HDIM, HDIM);
  k_gemm_fp8<<<gqkv, 256, 0, stream>>>(act8, acts, 0, 0, wv8, wvs, 0, 0, vb, 0, SEQ, HDIM, HDIM);

  k_rope<<<SEQ, 256, 0, stream>>>(qb, cosT, sinT);
  k_rope<<<SEQ, 256, 0, stream>>>(kb, cosT, sinT);

  k_quant_heads<<<SEQ, 256, 0, stream>>>(qb, qq8, qqs);
  k_quant_heads<<<SEQ, 256, 0, stream>>>(kb, kk8, kks);
  k_quant_vT<<<NHEAD * DHEAD, 256, 0, stream>>>(vb, vt8, vts);

  dim3 gqk(SEQ / 128, SEQ / 128, NHEAD);
  k_gemm_fp8<<<gqk, 256, 0, stream>>>(qq8, qqs, (long long)SEQ * DHEAD, (long long)SEQ,
                                      kk8, kks, (long long)SEQ * DHEAD, (long long)SEQ,
                                      scores, (long long)SEQ * SEQ, SEQ, SEQ, DHEAD);

  dim3 gsm(SEQ, NHEAD);
  k_softmax_quant<<<gsm, 256, 0, stream>>>(scores, p8, ps);

  dim3 gpv(SEQ / 128, DHEAD / 128, NHEAD);
  k_gemm_fp8<<<gpv, 256, 0, stream>>>(p8, ps, (long long)SEQ * SEQ, (long long)SEQ * (SEQ / 128),
                                      vt8, vts, (long long)DHEAD * SEQ, (long long)DHEAD * (SEQ / 128),
                                      ao, (long long)SEQ * DHEAD, SEQ, DHEAD, SEQ);

  k_quant_from_heads<<<SEQ, 256, 0, stream>>>(ao, act8, acts);
  k_gemm_fp8<<<gqkv, 256, 0, stream>>>(act8, acts, 0, 0, wo8, wos, 0, 0, aout, 0, SEQ, HDIM, HDIM);
  k_add<<<2048, 256, 0, stream>>>(hidden, aout, h1, (size_t)SEQ * HDIM);

  k_rmsnorm<<<SEQ, 256, 0, stream>>>(h1, ln2w, hn, HDIM);
  k_quant_rows<<<SEQ, 256, 0, stream>>>(hn, act8, acts, HDIM);
  dim3 gff(SEQ / 128, FFDIM / 128, 1);
  k_gemm_fp8<<<gff, 256, 0, stream>>>(act8, acts, 0, 0, wg8, wgs, 0, 0, gb_, 0, SEQ, FFDIM, HDIM);
  k_gemm_fp8<<<gff, 256, 0, stream>>>(act8, acts, 0, 0, wu8, wus, 0, 0, ub_, 0, SEQ, FFDIM, HDIM);
  k_silu_mul<<<2048, 256, 0, stream>>>(gb_, ub_, mb_, (size_t)SEQ * FFDIM);
  k_quant_rows<<<SEQ, 256, 0, stream>>>(mb_, aff8, affs, FFDIM);
  dim3 gd(SEQ / 128, HDIM / 128, 1);
  k_gemm_fp8<<<gd, 256, 0, stream>>>(aff8, affs, 0, 0, wd8, wds, 0, 0, db_, 0, SEQ, HDIM, FFDIM);
  k_add<<<2048, 256, 0, stream>>>(h1, db_, outp, (size_t)SEQ * HDIM);
}
